// TitansMemoryNode_16286515987260
// MI455X (gfx1250) — matrix operand order repaired and checked by reading, not compiled, not run on hardware
//
#include <hip/hip_runtime.h>
#include <hip/hip_bf16.h>
#include <math.h>

typedef __bf16 bf16_t;
typedef __attribute__((ext_vector_type(16))) __bf16 v16bf;
typedef __attribute__((ext_vector_type(8)))  __bf16 v8bf;
typedef __attribute__((ext_vector_type(8)))  float  v8f;

#define NB 8
#define NT 512
#define ND 1024
#define NF 64

// ---------------------------------------------------------------- utilities
__global__ void __launch_bounds__(256)
cvt_bf16_kernel(const float* __restrict__ s, bf16_t* __restrict__ d, int n) {
  int i = blockIdx.x * 256 + threadIdx.x;
  if (i < n) d[i] = (bf16_t)s[i];
}

// One block per (b,t) row: RMSNorm over D, emit bf16 x_norm.
__global__ void __launch_bounds__(256)
rmsnorm_kernel(const float* __restrict__ x, const float* __restrict__ w,
               bf16_t* __restrict__ xn) {
  long row = blockIdx.x;
  const float* xr = x + row * ND;
  float v[4];
  float ss = 0.f;
#pragma unroll
  for (int i = 0; i < 4; ++i) {
    int d = threadIdx.x + i * 256;
    v[i] = xr[d];
    ss += v[i] * v[i];
  }
#pragma unroll
  for (int o = 16; o > 0; o >>= 1) ss += __shfl_xor(ss, o, 32);
  __shared__ float sred[8];
  if ((threadIdx.x & 31) == 0) sred[threadIdx.x >> 5] = ss;
  __syncthreads();
  float tot = 0.f;
#pragma unroll
  for (int j = 0; j < 8; ++j) tot += sred[j];
  float rstd = rsqrtf(tot * (1.0f / ND) + 1e-6f);
#pragma unroll
  for (int i = 0; i < 4; ++i) {
    int d = threadIdx.x + i * 256;
    xn[row * ND + d] = (bf16_t)(v[i] * rstd * w[d]);
  }
}

// ---------------------------------------------------------------- WMMA GEMM
// C[M x N] = A[M x K] * W[N x K]^T   (bf16 in, f32 accumulate)
// Each wave: one 16-row x 64-col strip, K-loop in steps of 32 via
// v_wmma_f32_16x16x32_bf16.  epi: 0=none, 1=phi(elu+1), 2=residual add (R).
__global__ void __launch_bounds__(256)
gemm_bf16_wmma(const bf16_t* __restrict__ A, const bf16_t* __restrict__ W,
               float* __restrict__ Cf, bf16_t* __restrict__ Cb,
               const float* __restrict__ R,
               int lda, int ldw, int ldc, int K,
               long sA, long sW, long sC, int epi) {
  const int lane = threadIdx.x & 31;
  const int wave = threadIdx.x >> 5;
  const int row0 = (blockIdx.y * 8 + wave) * 16;
  const int col0 = blockIdx.x * 64;
  const long bz = blockIdx.z;
  A += bz * sA;
  W += bz * sW;
  const long cbase = bz * sC;

  const int mr   = lane & 15;   // M row (A) / N col (B) within tile
  const int half = lane >> 4;   // lane-half selects K sub-block

  // A fragment source: 16 contiguous K per lane, +16 for upper half
  const bf16_t* arow = A + (long)(row0 + mr) * lda + half * 16;
  // B fragment source: 16 contiguous K per lane, +16 for upper half
  const bf16_t* wp0 = W + (long)(col0 +  0 + mr) * ldw + half * 16;
  const bf16_t* wp1 = W + (long)(col0 + 16 + mr) * ldw + half * 16;
  const bf16_t* wp2 = W + (long)(col0 + 32 + mr) * ldw + half * 16;
  const bf16_t* wp3 = W + (long)(col0 + 48 + mr) * ldw + half * 16;

  v8f acc0 = {}; v8f acc1 = {}; v8f acc2 = {}; v8f acc3 = {};

  for (int kk = 0; kk < K; kk += 32) {
    v8bf lo = *(const v8bf*)(arow + kk);
    v8bf hi = *(const v8bf*)(arow + kk + 8);
    v16bf a;
#pragma unroll
    for (int e = 0; e < 8; ++e) { a[e] = lo[e]; a[e + 8] = hi[e]; }
    v16bf b0 = *(const v16bf*)(wp0 + kk);
    v16bf b1 = *(const v16bf*)(wp1 + kk);
    v16bf b2 = *(const v16bf*)(wp2 + kk);
    v16bf b3 = *(const v16bf*)(wp3 + kk);
    acc0 = __builtin_amdgcn_wmma_f32_16x16x32_bf16(false, a, false, b0, (short)0, acc0, false, false);
    acc1 = __builtin_amdgcn_wmma_f32_16x16x32_bf16(false, a, false, b1, (short)0, acc1, false, false);
    acc2 = __builtin_amdgcn_wmma_f32_16x16x32_bf16(false, a, false, b2, (short)0, acc2, false, false);
    acc3 = __builtin_amdgcn_wmma_f32_16x16x32_bf16(false, a, false, b3, (short)0, acc3, false, false);
  }

  const int n   = mr;
  const int mi0 = half * 8;
  v8f accs[4] = {acc0, acc1, acc2, acc3};
#pragma unroll
  for (int c = 0; c < 4; ++c) {
    const int col = col0 + c * 16 + n;
#pragma unroll
    for (int i = 0; i < 8; ++i) {
      const int row = row0 + mi0 + i;
      float vv = accs[c][i];
      if (epi == 1)      vv = (vv > 0.f) ? (vv + 1.f) : __expf(vv);
      else if (epi == 2) vv += R[(long)row * ldc + col];
      const long idx = cbase + (long)row * ldc + col;
      if (Cb) Cb[idx] = (bf16_t)vv;
      else    Cf[idx] = vv;
    }
  }
}

// ---------------------------------------------------------------- scan
__device__ __forceinline__ float wave_sum(float v) {
#pragma unroll
  for (int o = 16; o > 0; o >>= 1) v += __shfl_xor(v, o, 32);
  return v;
}

// One block per batch.  M (64x1024 f32 = 256 KB) lives entirely in the
// 320 KB WGP LDS for all 512 sequential steps.  Thread d owns column d.
__global__ void __launch_bounds__(1024)
scan_kernel(const float* __restrict__ kmat, const float* __restrict__ vmat,
            const float* __restrict__ eta, float* __restrict__ Mout,
            bf16_t* __restrict__ Mt) {
  extern __shared__ float Msh[];           // NF*ND floats (dynamic, 256 KB)
  __shared__ float s_ks[NF];
  __shared__ float s_eta[NF];
  __shared__ float s_red[64];
  __shared__ float s_bc[2];

  const int b = blockIdx.x;
  const int d = threadIdx.x;
  const int lane = d & 31;
  const int wid  = d >> 5;

#pragma unroll 4
  for (int f = 0; f < NF; ++f) Msh[f * ND + d] = 0.f;
  if (d < NF) s_eta[d] = fmaxf(fabsf(eta[d]), 1e-6f);
  __syncthreads();

#pragma unroll 1
  for (int t = 0; t < NT; ++t) {
    if (d < NF) s_ks[d] = kmat[((long)(b * NT + t)) * NF + d];
    __syncthreads();
    const float vt = vmat[((long)(b * NT + t)) * ND + d];

    // v_pred[d] = sum_f k[f] * M[f,d]   (own column only)
    float vp = 0.f;
#pragma unroll 8
    for (int f = 0; f < NF; ++f) vp += s_ks[f] * Msh[f * ND + d];
    const float diff = vt - vp;

    // surprise = || v - v_pred ||
    float ss = wave_sum(diff * diff);
    if (lane == 0) s_red[wid] = ss;
    __syncthreads();
    if (wid == 0) {
      float r = wave_sum(s_red[lane]);
      if (lane == 0) s_bc[0] = sqrtf(r);
    }
    __syncthreads();
    const float surprise = s_bc[0];

    // M += eta * surprise * (k ⊗ v); accumulate LN statistics in same pass
    float s1 = 0.f, s2 = 0.f;
#pragma unroll 8
    for (int f = 0; f < NF; ++f) {
      float m = Msh[f * ND + d] + (s_eta[f] * surprise) * s_ks[f] * vt;
      Msh[f * ND + d] = m;
      s1 += m;
      s2 += m * m;
    }
    s1 = wave_sum(s1);
    s2 = wave_sum(s2);
    if (lane == 0) { s_red[wid] = s1; s_red[32 + wid] = s2; }
    __syncthreads();
    if (wid == 0) {
      float r1 = wave_sum(s_red[lane]);
      float r2 = wave_sum(s_red[32 + lane]);
      if (lane == 0) {
        float mu  = r1 * (1.f / (NF * ND));
        float var = r2 * (1.f / (NF * ND)) - mu * mu;
        s_bc[0] = mu;
        s_bc[1] = rsqrtf(var + 1e-5f);
      }
    }
    __syncthreads();
    const float mu   = s_bc[0];
    const float rstd = s_bc[1];
#pragma unroll 8
    for (int f = 0; f < NF; ++f)
      Msh[f * ND + d] = (Msh[f * ND + d] - mu) * rstd;
    __syncthreads();
  }

  // outputs: final M (f32, part 2 of d_out) and transposed bf16 copy for WMMA
  const long mb = (long)b * NF * ND;
#pragma unroll 4
  for (int f = 0; f < NF; ++f) Mout[mb + f * ND + d] = Msh[f * ND + d];
  bf16_t* mt = Mt + ((long)b * ND + d) * NF;
#pragma unroll 4
  for (int f = 0; f < NF; ++f) mt[f] = (bf16_t)Msh[f * ND + d];
}

// ---------------------------------------------------------------- driver
extern "C" void kernel_launch(void* const* d_in, const int* in_sizes, int n_in,
                              void* d_out, int out_size, void* d_ws, size_t ws_size,
                              hipStream_t stream) {
  (void)in_sizes; (void)n_in; (void)out_size; (void)ws_size;
  const float* x    = (const float*)d_in[0];
  const float* nw   = (const float*)d_in[1];
  const float* Wk   = (const float*)d_in[2];
  const float* Wv   = (const float*)d_in[3];
  const float* Wq   = (const float*)d_in[4];
  const float* eta  = (const float*)d_in[5];
  const float* Wout = (const float*)d_in[6];
  float* out  = (float*)d_out;                    // (B,T,D) f32
  float* Mout = out + (long)NB * NT * ND;         // (B,F,D) f32

  char* w8 = (char*)d_ws;
  size_t off = 0;
  auto alloc = [&](size_t bytes) -> void* {
    void* p = w8 + off;
    off += (bytes + 255) & ~(size_t)255;
    return p;
  };
  bf16_t* Xn   = (bf16_t*)alloc((size_t)NB * NT * ND * 2);  // x_norm bf16
  bf16_t* Wkb  = (bf16_t*)alloc((size_t)NF * ND * 2);
  bf16_t* Wqb  = (bf16_t*)alloc((size_t)NF * ND * 2);
  bf16_t* Wvb  = (bf16_t*)alloc((size_t)ND * ND * 2);
  bf16_t* Wob  = (bf16_t*)alloc((size_t)ND * ND * 2);
  float*  kbuf = (float*) alloc((size_t)NB * NT * NF * 4);  // k (f32 for scan)
  bf16_t* qbuf = (bf16_t*)alloc((size_t)NB * NT * NF * 2);  // q (bf16 for GEMM)
  float*  vbuf = (float*) alloc((size_t)NB * NT * ND * 4);  // v (f32 for scan)
  bf16_t* mo   = (bf16_t*)alloc((size_t)NB * NT * ND * 2);  // mem_out bf16
  bf16_t* Mt   = (bf16_t*)alloc((size_t)NB * ND * NF * 2);  // M^T bf16

  cvt_bf16_kernel<<<dim3((NF * ND + 255) / 256), 256, 0, stream>>>(Wk, Wkb, NF * ND);
  cvt_bf16_kernel<<<dim3((NF * ND + 255) / 256), 256, 0, stream>>>(Wq, Wqb, NF * ND);
  cvt_bf16_kernel<<<dim3((ND * ND + 255) / 256), 256, 0, stream>>>(Wv, Wvb, ND * ND);
  cvt_bf16_kernel<<<dim3((ND * ND + 255) / 256), 256, 0, stream>>>(Wout, Wob, ND * ND);

  rmsnorm_kernel<<<dim3(NB * NT), 256, 0, stream>>>(x, nw, Xn);

  // k = phi(Xn Wk^T)  -> f32 (scan input)
  gemm_bf16_wmma<<<dim3(1, (NB * NT) / 128, 1), 256, 0, stream>>>(
      Xn, Wkb, kbuf, nullptr, nullptr, ND, ND, NF, ND, 0, 0, 0, 1);
  // q = phi(Xn Wq^T)  -> bf16 (GEMM input)
  gemm_bf16_wmma<<<dim3(1, (NB * NT) / 128, 1), 256, 0, stream>>>(
      Xn, Wqb, nullptr, qbuf, nullptr, ND, ND, NF, ND, 0, 0, 0, 1);
  // v = Xn Wv^T       -> f32 (scan input)
  gemm_bf16_wmma<<<dim3(ND / 64, (NB * NT) / 128, 1), 256, 0, stream>>>(
      Xn, Wvb, vbuf, nullptr, nullptr, ND, ND, ND, ND, 0, 0, 0, 0);

  // sequential memory scan, M resident in 256 KB of the 320 KB WGP LDS
  scan_kernel<<<dim3(NB), dim3(ND), (size_t)NF * ND * sizeof(float), stream>>>(
      kbuf, vbuf, eta, Mout, Mt);

  // mem_out[b] = q[b] (T x F) * Mt[b] (D x F)^T  -> bf16
  gemm_bf16_wmma<<<dim3(ND / 64, NT / 128, NB), 256, 0, stream>>>(
      qbuf, Mt, nullptr, mo, nullptr, NF, NF, ND, NF,
      (long)NT * NF, (long)ND * NF, (long)NT * ND, 0);

  // output = x + mem_out Wout^T  -> f32
  gemm_bf16_wmma<<<dim3(ND / 64, (NB * NT) / 128, 1), 256, 0, stream>>>(
      mo, Wob, out, nullptr, x, ND, ND, ND, ND, 0, 0, 0, 2);
}
